// EdgewiseReduce_49632642072860
// MI455X (gfx1250) — matrix operand order repaired and checked by reading, not compiled, not run on hardware
//
#include <hip/hip_runtime.h>

#define E_TOT 160000
#define NODES 10000
#define FDIM  288
#define EPB   32          // edges per block in the fused K kernel

typedef __bf16 bf16_t;
typedef __attribute__((ext_vector_type(16))) __bf16 v16bf;
typedef __attribute__((ext_vector_type(8)))  __bf16 v8bf;
typedef __attribute__((ext_vector_type(8)))  float  v8f;

__device__ __forceinline__ float fast_rcp(float x) { return __builtin_amdgcn_rcpf(x); }
__device__ __forceinline__ float silu(float x) {
  return x * fast_rcp(1.0f + __expf(-x));        // x * sigmoid(x)
}

__device__ __forceinline__ void atomicMaxF(float* a, float v) {
  // mixed int/uint ordering trick for IEEE754 float max
  if (v >= 0.0f) atomicMax((int*)a, __float_as_int(v));
  else           atomicMin((unsigned int*)a, __float_as_uint(v));
}

__device__ __forceinline__ v8f wmma_bf16(v16bf a, v16bf b, v8f c) {
  // D = A(16x32) * B(32x16) + C(16x16), f32 accumulate
  return __builtin_amdgcn_wmma_f32_16x16x32_bf16(false, a, false, b, (short)0, c, false, false);
}

// A fragment: 16x32 bf16 tile, row-major in LDS with stride ldb.
// Per lane this is two contiguous 16B runs -> 2x ds_load_b128.
__device__ __forceinline__ v16bf load_A_lds(const bf16_t* buf, int ldb, int lane) {
  const int row = lane & 15;
  const int khi = (lane >> 4) ? 8 : 0;
  v16bf a;
#pragma unroll
  for (int v = 0; v < 8; ++v) {
    const int kb = ((v < 4) ? (2 * v) : (16 + 2 * (v - 4))) + khi;
    a[2 * v]     = buf[row * ldb + kb];
    a[2 * v + 1] = buf[row * ldb + kb + 1];
  }
  return a;
}

// B fragment from pre-swizzled weights: each lane's 16 bf16 are contiguous
// (32B) -> 2x global_load_b128, perfectly coalesced, no packing movs.
__device__ __forceinline__ v16bf load_B_frag(const bf16_t* __restrict__ Wsw,
                                             int numKc, int kc, int nt, int lane) {
  return *(const v16bf*)(Wsw + (size_t)(((nt * numKc) + kc) * 32 + lane) * 16);
}

// ---------------------------------------------------------------------------
// weight f32 -> bf16 conversion + swizzle into WMMA B-fragment order.
// Element (k,n) -> frag (kc=k>>5, nt=n>>4), lane = (n&15)+((k&8)<<1), slot = (k&7)+((k&16)>>1).
// ---------------------------------------------------------------------------
__global__ void __launch_bounds__(256) wcvt_swz_kernel(const float* __restrict__ src,
                                                       bf16_t* __restrict__ dst,
                                                       int K, int N) {
  const int id = blockIdx.x * 256 + threadIdx.x;
  if (id >= K * N) return;
  const int k = id / N, n = id % N;
  const int numKc = K >> 5;
  const int kc = k >> 5;
  const int nt = n >> 4;
  const int lane = (n & 15) + ((k & 8) << 1);
  const int j = (k & 7) + ((k & 16) >> 1);
  dst[(size_t)(((nt * numKc) + kc) * 32 + lane) * 16 + j] = (bf16_t)src[id];
}

// ---------------------------------------------------------------------------
// init: zero output + z, set per-(node,channel) running max to -inf
// ---------------------------------------------------------------------------
__global__ void __launch_bounds__(256) init_kernel(float* __restrict__ out,
                                                   float* __restrict__ mbuf,
                                                   float* __restrict__ zbuf) {
  const int id = blockIdx.x * 256 + threadIdx.x;
  if (id < NODES * FDIM) out[id] = 0.0f;
  if (id < NODES * 32) { mbuf[id] = -__builtin_inff(); zbuf[id] = 0.0f; }
}

// ---------------------------------------------------------------------------
// Q-MLP per NODE (16x fewer rows than per-edge; gathered later):
//   Q = (silu(silu(attrs@W0+b0)@W1+b1))@W2+b2  -> Qbuf [NODES, 1024] bf16
// Block: 128 thr = 4 waves, 16 nodes/block, WMMA bf16.
// ---------------------------------------------------------------------------
__global__ void __launch_bounds__(128) q_mlp_kernel(
    const float* __restrict__ attrs,
    const bf16_t* __restrict__ W0, const float* __restrict__ b0,
    const bf16_t* __restrict__ W1, const float* __restrict__ b1,
    const bf16_t* __restrict__ W2, const float* __restrict__ b2,
    bf16_t* __restrict__ Qbuf)
{
  __shared__ bf16_t X[16 * 64];
  __shared__ bf16_t H[16 * 64];
  const int tid = threadIdx.x, wave = tid >> 5, lane = tid & 31;
  const int node0 = blockIdx.x * 16;
  const int col = lane & 15, moff = (lane >> 4) ? 8 : 0;

  for (int i = tid; i < 16 * 64; i += 128)
    X[i] = (bf16_t)attrs[(node0 + (i >> 6)) * 64 + (i & 63)];
  __syncthreads();

  { // layer 0: X[16,64] @ W0[64,64] -> H (silu)
    const int n0 = wave * 16;
    v8f acc = {};
#pragma unroll
    for (int kc = 0; kc < 2; ++kc)
      acc = wmma_bf16(load_A_lds(X + kc * 32, 64, lane), load_B_frag(W0, 2, kc, wave, lane), acc);
    const float bias = b0[n0 + col];
#pragma unroll
    for (int r = 0; r < 8; ++r)
      H[(r + moff) * 64 + n0 + col] = (bf16_t)silu(acc[r] + bias);
  }
  __syncthreads();
  { // layer 1: H @ W1[64,64] -> X (silu)
    const int n0 = wave * 16;
    v8f acc = {};
#pragma unroll
    for (int kc = 0; kc < 2; ++kc)
      acc = wmma_bf16(load_A_lds(H + kc * 32, 64, lane), load_B_frag(W1, 2, kc, wave, lane), acc);
    const float bias = b1[n0 + col];
#pragma unroll
    for (int r = 0; r < 8; ++r)
      X[(r + moff) * 64 + n0 + col] = (bf16_t)silu(acc[r] + bias);
  }
  __syncthreads();
  // layer 2: X[16,64] @ W2[64,1024] -> Qbuf bf16 (no activation). A hoisted.
  {
    const v16bf a0 = load_A_lds(X, 64, lane);
    const v16bf a1 = load_A_lds(X + 32, 64, lane);
    for (int t = 0; t < 16; ++t) {
      const int nt = t * 4 + wave, n0 = nt * 16;
      v8f acc = {};
      acc = wmma_bf16(a0, load_B_frag(W2, 2, 0, nt, lane), acc);
      acc = wmma_bf16(a1, load_B_frag(W2, 2, 1, nt, lane), acc);
      const float bias = b2[n0 + col];
#pragma unroll
      for (int r = 0; r < 8; ++r)
        Qbuf[(size_t)(node0 + r + moff) * 1024 + n0 + col] = (bf16_t)(acc[r] + bias);
    }
  }
}

// ---------------------------------------------------------------------------
// K-MLP per EDGE fused with Q-gather dot + logits + segment max.
// 32 edges/block, 2 row tiles per wave share each B fragment.
// K tile lives in LDS (bf16, 64KB); never hits HBM.
//   Wlog[e,c] = 5 * sum_d K[e,c,d] * Q[center(e),c,d];  mbuf = segment max.
// ---------------------------------------------------------------------------
__global__ void __launch_bounds__(128) k_mlp_logits_kernel(
    const float* __restrict__ edge_feat, const int* __restrict__ edge_center,
    const bf16_t* __restrict__ Qbuf,
    const bf16_t* __restrict__ W0, const float* __restrict__ b0,
    const bf16_t* __restrict__ W1, const float* __restrict__ b1,
    const bf16_t* __restrict__ W2, const float* __restrict__ b2,
    float* __restrict__ Wlog, float* __restrict__ mbuf)
{
  __shared__ bf16_t X[EPB * 32];
  __shared__ bf16_t H1[EPB * 64];
  __shared__ bf16_t H2[EPB * 64];
  __shared__ bf16_t Kb[EPB * 1024];
  __shared__ int    nid[EPB];
  const int tid = threadIdx.x, wave = tid >> 5, lane = tid & 31;
  const int e0 = blockIdx.x * EPB;
  const int col = lane & 15, moff = (lane >> 4) ? 8 : 0;

  if (tid < EPB) nid[tid] = edge_center[e0 + tid];
  for (int i = tid; i < EPB * 32; i += 128)
    X[i] = (bf16_t)edge_feat[(size_t)(e0 + (i >> 5)) * FDIM + (i & 31)];
  __syncthreads();

  { // layer 0: X[32,32] @ W0[32,64] -> H1 (silu); one B per wave, 2 row tiles
    const int n0 = wave * 16;
    const v16bf b = load_B_frag(W0, 1, 0, wave, lane);
    v8f acc0 = {}, acc1 = {};
    acc0 = wmma_bf16(load_A_lds(X, 32, lane), b, acc0);
    acc1 = wmma_bf16(load_A_lds(X + 16 * 32, 32, lane), b, acc1);
    const float bias = b0[n0 + col];
#pragma unroll
    for (int r = 0; r < 8; ++r) {
      H1[(r + moff) * 64 + n0 + col]      = (bf16_t)silu(acc0[r] + bias);
      H1[(16 + r + moff) * 64 + n0 + col] = (bf16_t)silu(acc1[r] + bias);
    }
  }
  __syncthreads();
  { // layer 1: H1[32,64] @ W1[64,64] -> H2 (silu)
    const int n0 = wave * 16;
    v8f acc0 = {}, acc1 = {};
#pragma unroll
    for (int kc = 0; kc < 2; ++kc) {
      const v16bf b = load_B_frag(W1, 2, kc, wave, lane);
      acc0 = wmma_bf16(load_A_lds(H1 + kc * 32, 64, lane), b, acc0);
      acc1 = wmma_bf16(load_A_lds(H1 + 16 * 64 + kc * 32, 64, lane), b, acc1);
    }
    const float bias = b1[n0 + col];
#pragma unroll
    for (int r = 0; r < 8; ++r) {
      H2[(r + moff) * 64 + n0 + col]      = (bf16_t)silu(acc0[r] + bias);
      H2[(16 + r + moff) * 64 + n0 + col] = (bf16_t)silu(acc1[r] + bias);
    }
  }
  __syncthreads();
  // layer 2: H2[32,64] @ W2[64,1024] -> Kb in LDS (bf16, +bias).
  // A fragments are t-invariant: hoist all 4 into registers.
  {
    const v16bf a00 = load_A_lds(H2, 64, lane);
    const v16bf a01 = load_A_lds(H2 + 32, 64, lane);
    const v16bf a10 = load_A_lds(H2 + 16 * 64, 64, lane);
    const v16bf a11 = load_A_lds(H2 + 16 * 64 + 32, 64, lane);
    for (int t = 0; t < 16; ++t) {
      const int nt = t * 4 + wave, n0 = nt * 16;
      const v16bf b0f = load_B_frag(W2, 2, 0, nt, lane);
      const v16bf b1f = load_B_frag(W2, 2, 1, nt, lane);
      v8f acc0 = {}, acc1 = {};
      acc0 = wmma_bf16(a00, b0f, acc0);
      acc0 = wmma_bf16(a01, b1f, acc0);
      acc1 = wmma_bf16(a10, b0f, acc1);
      acc1 = wmma_bf16(a11, b1f, acc1);
      const float bias = b2[n0 + col];
#pragma unroll
      for (int r = 0; r < 8; ++r) {
        Kb[(r + moff) * 1024 + n0 + col]      = (bf16_t)(acc0[r] + bias);
        Kb[(16 + r + moff) * 1024 + n0 + col] = (bf16_t)(acc1[r] + bias);
      }
    }
  }
  __syncthreads();

  // logits + segment max: 32 edges x 32 channels = 1024 dots, 8 per thread
  for (int p = tid; p < EPB * 32; p += 128) {
    const int e = p >> 5, c = p & 31;
    const int ge = e0 + e;
    const int node = nid[e];
    const v8bf* q8 = (const v8bf*)(Qbuf + (size_t)node * 1024 + c * 32);
    const v8bf* k8 = (const v8bf*)(Kb + e * 1024 + c * 32);
    float s = 0.0f;
#pragma unroll
    for (int i = 0; i < 4; ++i) {
      const v8bf kv = k8[i], qv = q8[i];
#pragma unroll
      for (int j = 0; j < 8; ++j) s += (float)kv[j] * (float)qv[j];
    }
    s *= 5.0f;                       // ISQRTD
    Wlog[(size_t)ge * 32 + c] = s;
    atomicMaxF(&mbuf[node * 32 + c], s);
  }
}

// ---------------------------------------------------------------------------
// pass 2 of segment softmax: e = exp(w - max[seg]); Wlog := e; z[seg] += e
// ---------------------------------------------------------------------------
__global__ void __launch_bounds__(256) softmax_z_kernel(
    const int* __restrict__ edge_center, float* __restrict__ Wlog,
    const float* __restrict__ mbuf, float* __restrict__ zbuf)
{
  const int id = blockIdx.x * 256 + threadIdx.x;       // grid covers E*32 exactly
  const int e = id >> 5, c = id & 31;
  const int node = edge_center[e];
  const float v = __expf(Wlog[id] - mbuf[node * 32 + c]);
  Wlog[id] = v;
  atomicAdd(&zbuf[node * 32 + c], v);
}

// ---------------------------------------------------------------------------
// pass 3: attn = e / z[seg]; scale 288 cols by their channel's attn and
// scatter-add into out[node]. One wave per edge, 9 cols per lane.
// Column->channel map: j<32: c=j; 32<=j<128: c=(j-32)/3; j>=128: c=(j-128)/5.
// ---------------------------------------------------------------------------
__global__ void __launch_bounds__(256) scatter_kernel(
    const float* __restrict__ edge_feat, const int* __restrict__ edge_center,
    const float* __restrict__ Wexp, const float* __restrict__ zbuf,
    float* __restrict__ out)
{
  __shared__ float attn[8 * 32];
  const int w = threadIdx.x >> 5, lane = threadIdx.x & 31;
  const int e = blockIdx.x * 8 + w;                    // grid covers E exactly
  const int node = edge_center[e];
  attn[w * 32 + lane] = Wexp[(size_t)e * 32 + lane] * fast_rcp(zbuf[node * 32 + lane]);
  __syncthreads();
#pragma unroll
  for (int t = 0; t < 9; ++t) {
    const int j = t * 32 + lane;
    const int c = (j < 32) ? j : ((j < 128) ? ((j - 32) / 3) : ((j - 128) / 5));
    atomicAdd(&out[(size_t)node * FDIM + j],
              edge_feat[(size_t)e * FDIM + j] * attn[w * 32 + c]);
  }
}

extern "C" void kernel_launch(void* const* d_in, const int* in_sizes, int n_in,
                              void* d_out, int out_size, void* d_ws, size_t ws_size,
                              hipStream_t stream) {
  (void)in_sizes; (void)n_in; (void)out_size; (void)ws_size;
  const float* edge_feat   = (const float*)d_in[0];
  const float* node_attrs  = (const float*)d_in[1];
  const int*   edge_center = (const int*)d_in[2];
  const float* Wq0 = (const float*)d_in[3];  const float* bq0 = (const float*)d_in[4];
  const float* Wq1 = (const float*)d_in[5];  const float* bq1 = (const float*)d_in[6];
  const float* Wq2 = (const float*)d_in[7];  const float* bq2 = (const float*)d_in[8];
  const float* Wk0 = (const float*)d_in[9];  const float* bk0 = (const float*)d_in[10];
  const float* Wk1 = (const float*)d_in[11]; const float* bk1 = (const float*)d_in[12];
  const float* Wk2 = (const float*)d_in[13]; const float* bk2 = (const float*)d_in[14];
  float* out = (float*)d_out;

  // ws layout: Q bf16 [NODES,1024] | Wlog f32 [E,32] | m,z f32 [NODES,32] | swizzled bf16 weights
  bf16_t* Qbuf = (bf16_t*)d_ws;
  float*  Wlog = (float*)(Qbuf + (size_t)NODES * 1024);
  float*  mbuf = Wlog + (size_t)E_TOT * 32;
  float*  zbuf = mbuf + (size_t)NODES * 32;
  bf16_t* Wq0b = (bf16_t*)(zbuf + (size_t)NODES * 32);
  bf16_t* Wq1b = Wq0b + 64 * 64;
  bf16_t* Wq2b = Wq1b + 64 * 64;
  bf16_t* Wk0b = Wq2b + 64 * 1024;
  bf16_t* Wk1b = Wk0b + 32 * 64;
  bf16_t* Wk2b = Wk1b + 64 * 64;

  wcvt_swz_kernel<<<(64 * 64 + 255) / 256, 256, 0, stream>>>(Wq0, Wq0b, 64, 64);
  wcvt_swz_kernel<<<(64 * 64 + 255) / 256, 256, 0, stream>>>(Wq1, Wq1b, 64, 64);
  wcvt_swz_kernel<<<(64 * 1024 + 255) / 256, 256, 0, stream>>>(Wq2, Wq2b, 64, 1024);
  wcvt_swz_kernel<<<(32 * 64 + 255) / 256, 256, 0, stream>>>(Wk0, Wk0b, 32, 64);
  wcvt_swz_kernel<<<(64 * 64 + 255) / 256, 256, 0, stream>>>(Wk1, Wk1b, 64, 64);
  wcvt_swz_kernel<<<(64 * 1024 + 255) / 256, 256, 0, stream>>>(Wk2, Wk2b, 64, 1024);

  init_kernel<<<(NODES * FDIM) / 256, 256, 0, stream>>>(out, mbuf, zbuf);
  q_mlp_kernel<<<NODES / 16, 128, 0, stream>>>(node_attrs, Wq0b, bq0, Wq1b, bq1, Wq2b, bq2, Qbuf);
  k_mlp_logits_kernel<<<E_TOT / EPB, 128, 0, stream>>>(edge_feat, edge_center, Qbuf,
                                                       Wk0b, bk0, Wk1b, bk1, Wk2b, bk2,
                                                       Wlog, mbuf);
  softmax_z_kernel<<<(E_TOT * 32) / 256, 256, 0, stream>>>(edge_center, Wlog, mbuf, zbuf);
  scatter_kernel<<<E_TOT / 8, 256, 0, stream>>>(edge_feat, edge_center, Wlog, zbuf, out);
}
